// CFGKT_46531675685179
// MI455X (gfx1250) — compile-verified, matrix operand order checked by reading, not run on hardware
//
#include <hip/hip_runtime.h>

// ---------------------------------------------------------------------------
// CFGKT forward on MI455X (gfx1250): bf16 WMMA GEMMs + flash attention.
// B=8 L=1024 D=512 H=8 DK=64 DFF=2048, M = B*L = 8192 rows.
// ---------------------------------------------------------------------------

#define B_    8
#define L_    1024
#define D_    512
#define H_    8
#define DK_   64
#define DFF_  2048
#define M_    (B_ * L_)   // 8192

typedef float  f32x4  __attribute__((ext_vector_type(4)));
typedef float  v8f    __attribute__((ext_vector_type(8)));
typedef __bf16 v16bf  __attribute__((ext_vector_type(16)));
typedef __bf16 bf16x4 __attribute__((ext_vector_type(4)));

union FragAB { v16bf v; f32x4 f4[2]; };

static __device__ __forceinline__ v8f wmma_bf16(v16bf a, v16bf b, v8f c) {
  return __builtin_amdgcn_wmma_f32_16x16x32_bf16(false, a, false, b,
                                                 (short)0, c, false, false);
}

// ---------------------------------------------------------------------------
// GEMM: C[M,N] = A[M,K] @ W[K,N] + bias  (optional fused ReLU)
// 256 threads (8 waves), 128x128x32 tile, per-wave 64x32 (4x2 WMMA frags).
// A and W^T staged in LDS as bf16, row stride 40 (80B, 16B-aligned chunks).
// Register double-buffering: next K-tile's global loads issue before the
// fragment-load + WMMA phase so HBM latency hides behind the MMAs.
// ---------------------------------------------------------------------------
template <bool RELU>
__global__ __launch_bounds__(256)
void gemm_bias_kernel(const float* __restrict__ A, const float* __restrict__ W,
                      const float* __restrict__ bias, float* __restrict__ C,
                      int M, int N, int K) {
  __shared__ alignas(16) __bf16 a_s[128 * 40];
  __shared__ alignas(16) __bf16 b_s[128 * 40];
  (void)M;

  const int tid  = threadIdx.x;
  const int lane = tid & 31;
  const int wid  = tid >> 5;
  const int half = lane >> 4;
  const int ln16 = lane & 15;
  const int m0   = blockIdx.y * 128;
  const int n0   = blockIdx.x * 128;
  const int wm   = (wid >> 2) * 64;   // 2 waves along M
  const int wn   = (wid & 3) * 32;    // 4 waves along N

  const v8f vzero = {};
  v8f acc[4][2];
#pragma unroll
  for (int i = 0; i < 4; ++i)
#pragma unroll
    for (int j = 0; j < 2; ++j) acc[i][j] = vzero;

  const int ar   = tid >> 3;          // 0..31  A loader row
  const int acol = (tid & 7) * 4;     // 0..28  A loader col
  const int wr   = tid >> 5;          // 0..7   W loader k-row
  const int wc   = (tid & 31) * 4;    // 0..124 W loader n-col

  // Prologue: load K-tile 0 into registers.
  f32x4 areg[4], wreg[4];
#pragma unroll
  for (int it = 0; it < 4; ++it) {
    areg[it] = *(const f32x4*)(A + (size_t)(m0 + ar + 32 * it) * K + acol);
    wreg[it] = *(const f32x4*)(W + (size_t)(wr + 8 * it) * N + n0 + wc);
  }

  for (int k0 = 0; k0 < K; k0 += 32) {
    // Stage current tile to LDS (packed bf16 converts, b64 row stores).
#pragma unroll
    for (int it = 0; it < 4; ++it) {
      int r = ar + 32 * it;
      *(bf16x4*)(a_s + r * 40 + acol) = __builtin_convertvector(areg[it], bf16x4);
      int kr = wr + 8 * it;
      f32x4 wv = wreg[it];
      b_s[(wc + 0) * 40 + kr] = (__bf16)wv.x;
      b_s[(wc + 1) * 40 + kr] = (__bf16)wv.y;
      b_s[(wc + 2) * 40 + kr] = (__bf16)wv.z;
      b_s[(wc + 3) * 40 + kr] = (__bf16)wv.w;
    }
    __syncthreads();

    // Kick off next tile's global loads; they complete during the MMA phase.
    if (k0 + 32 < K) {
#pragma unroll
      for (int it = 0; it < 4; ++it) {
        areg[it] = *(const f32x4*)(A + (size_t)(m0 + ar + 32 * it) * K +
                                   (k0 + 32) + acol);
        wreg[it] = *(const f32x4*)(W + (size_t)(k0 + 32 + wr + 8 * it) * N +
                                   n0 + wc);
      }
    }

    FragAB afr[4], bfr[2];
#pragma unroll
    for (int mi = 0; mi < 4; ++mi) {
      const f32x4* p = (const f32x4*)(a_s + (wm + mi * 16 + ln16) * 40);
      afr[mi].f4[0] = p[half];
      afr[mi].f4[1] = p[2 + half];
    }
#pragma unroll
    for (int ni = 0; ni < 2; ++ni) {
      const f32x4* p = (const f32x4*)(b_s + (wn + ni * 16 + ln16) * 40);
      bfr[ni].f4[0] = p[half];
      bfr[ni].f4[1] = p[2 + half];
    }
#pragma unroll
    for (int mi = 0; mi < 4; ++mi)
#pragma unroll
      for (int ni = 0; ni < 2; ++ni)
        acc[mi][ni] = wmma_bf16(afr[mi].v, bfr[ni].v, acc[mi][ni]);
    __syncthreads();
  }

  // Epilogue: C/D layout -> lane ln16 = col, (half*8 + v) = row.
#pragma unroll
  for (int ni = 0; ni < 2; ++ni) {
    int n = n0 + wn + ni * 16 + ln16;
    float bv = bias[n];
#pragma unroll
    for (int mi = 0; mi < 4; ++mi) {
      int mb = m0 + wm + mi * 16 + half * 8;
#pragma unroll
      for (int v = 0; v < 8; ++v) {
        float x = acc[mi][ni][v] + bv;
        if (RELU) x = fmaxf(x, 0.0f);
        C[(size_t)(mb + v) * N + n] = x;
      }
    }
  }
}

// ---------------------------------------------------------------------------
// Flash attention per (b,h): out = softmax(mask(Q K^T / 8)) V
// Q==K (shared projection). grid=(L/64, B*H), 128 threads (4 waves x 16 rows).
// QKV live in [B,L,D] buffers at column h*64.  mask: key <= query + mask_k - 1.
// ---------------------------------------------------------------------------
__global__ __launch_bounds__(128)
void attn_kernel(const float* __restrict__ KQ, const float* __restrict__ V,
                 float* __restrict__ O, int mask_k) {
  __shared__ alignas(16) __bf16 q_s[64 * 72];
  __shared__ alignas(16) __bf16 k_s[32 * 72];
  __shared__ alignas(16) __bf16 vt_s[64 * 40];
  __shared__ alignas(16) __bf16 p_s[4 * 16 * 40];

  const int tid  = threadIdx.x;
  const int lane = tid & 31;
  const int wid  = tid >> 5;       // 0..3
  const int half = lane >> 4;
  const int ln16 = lane & 15;
  const int bh   = blockIdx.y;
  const int b    = bh >> 3;
  const int h    = bh & 7;
  const int q0   = blockIdx.x * 64;
  const size_t base = (size_t)b * L_ * D_ + (size_t)h * DK_;

  const int c4 = (tid & 15) * 4;
  const int r0 = tid >> 4;

  // Load Q tile (64 x 64), pre-scaled by 1/sqrt(64) = 0.125 (exact pow2).
#pragma unroll
  for (int it = 0; it < 8; ++it) {
    int r = r0 + it * 8;
    f32x4 qv = *(const f32x4*)(KQ + base + (size_t)(q0 + r) * D_ + c4);
    qv *= 0.125f;
    *(bf16x4*)(q_s + r * 72 + c4) = __builtin_convertvector(qv, bf16x4);
  }

  const v8f vzero = {};
  float rm[8], rs[8];
  v8f of[4];
#pragma unroll
  for (int v = 0; v < 8; ++v) { rm[v] = -1e30f; rs[v] = 0.0f; }
#pragma unroll
  for (int dn = 0; dn < 4; ++dn) of[dn] = vzero;

  const int mrow0 = q0 + wid * 16 + half * 8;    // + v -> global q row
  const int nkb = (q0 + 64) >> 5;                // causal: keys <= q0+63

  for (int kb = 0; kb < nkb; ++kb) {
    const int kk0 = kb * 32;
    __syncthreads();
#pragma unroll
    for (int it = 0; it < 4; ++it) {
      int r = r0 + it * 8;
      f32x4 kv = *(const f32x4*)(KQ + base + (size_t)(kk0 + r) * D_ + c4);
      *(bf16x4*)(k_s + r * 72 + c4) = __builtin_convertvector(kv, bf16x4);
    }
#pragma unroll
    for (int it = 0; it < 4; ++it) {
      int r = r0 + it * 8;
      f32x4 vv = *(const f32x4*)(V + base + (size_t)(kk0 + r) * D_ + c4);
      vt_s[(c4 + 0) * 40 + r] = (__bf16)vv.x;
      vt_s[(c4 + 1) * 40 + r] = (__bf16)vv.y;
      vt_s[(c4 + 2) * 40 + r] = (__bf16)vv.z;
      vt_s[(c4 + 3) * 40 + r] = (__bf16)vv.w;
    }
    // Prefetch next key block while this one is consumed (global_prefetch_b8).
    if (kb + 1 < nkb) {
      __builtin_prefetch(KQ + base + (size_t)(kk0 + 32 + r0) * D_ + c4, 0, 1);
      __builtin_prefetch(V + base + (size_t)(kk0 + 32 + r0) * D_ + c4, 0, 1);
    }
    __syncthreads();

    // S = Q K^T for this wave's 16 rows x 32 key cols (2 k-steps of 32 dims)
    v8f sf[2];
    sf[0] = vzero; sf[1] = vzero;
#pragma unroll
    for (int ks = 0; ks < 2; ++ks) {
      FragAB qf;
      const f32x4* qp = (const f32x4*)(q_s + (wid * 16 + ln16) * 72);
      qf.f4[0] = qp[4 * ks + half];
      qf.f4[1] = qp[4 * ks + 2 + half];
#pragma unroll
      for (int ni = 0; ni < 2; ++ni) {
        FragAB kf;
        const f32x4* kp = (const f32x4*)(k_s + (ni * 16 + ln16) * 72);
        kf.f4[0] = kp[4 * ks + half];
        kf.f4[1] = kp[4 * ks + 2 + half];
        sf[ni] = wmma_bf16(qf.v, kf.v, sf[ni]);
      }
    }

    // Causal mask: allow key n <= row m + mask_k - 1
#pragma unroll
    for (int ni = 0; ni < 2; ++ni) {
      int n_g = kk0 + ni * 16 + ln16;
#pragma unroll
      for (int v = 0; v < 8; ++v)
        if (n_g > mrow0 + v + (mask_k - 1)) sf[ni][v] = -1e30f;
    }

    // Online softmax: row max over the 16-lane half (rows are per (half,v)).
    float alpha[8];
#pragma unroll
    for (int v = 0; v < 8; ++v) {
      float mx = fmaxf(sf[0][v], sf[1][v]);
#pragma unroll
      for (int s = 1; s < 16; s <<= 1) mx = fmaxf(mx, __shfl_xor(mx, s, 32));
      float mnew = fmaxf(rm[v], mx);
      alpha[v] = __expf(rm[v] - mnew);
      rm[v] = mnew;
    }

    // P = exp(S - m): keep f32 for row sums, bounce bf16 via LDS to A-layout.
#pragma unroll
    for (int ni = 0; ni < 2; ++ni)
#pragma unroll
      for (int v = 0; v < 8; ++v) {
        float p = __expf(sf[ni][v] - rm[v]);
        sf[ni][v] = p;
        p_s[wid * 640 + (half * 8 + v) * 40 + ni * 16 + ln16] = (__bf16)p;
      }
#pragma unroll
    for (int v = 0; v < 8; ++v) {
      float ps = sf[0][v] + sf[1][v];
#pragma unroll
      for (int s = 1; s < 16; s <<= 1) ps += __shfl_xor(ps, s, 32);
      rs[v] = rs[v] * alpha[v] + ps;
    }

    // Rescale O then accumulate P @ V  (V^T in LDS gives B-fragments).
#pragma unroll
    for (int dn = 0; dn < 4; ++dn)
#pragma unroll
      for (int v = 0; v < 8; ++v) of[dn][v] *= alpha[v];

    FragAB pf;
    const f32x4* pp = (const f32x4*)(p_s + wid * 640 + ln16 * 40);
    pf.f4[0] = pp[half];
    pf.f4[1] = pp[2 + half];
#pragma unroll
    for (int dn = 0; dn < 4; ++dn) {
      FragAB vf;
      const f32x4* vp = (const f32x4*)(vt_s + (dn * 16 + ln16) * 40);
      vf.f4[0] = vp[half];
      vf.f4[1] = vp[2 + half];
      of[dn] = wmma_bf16(pf.v, vf.v, of[dn]);
    }
  }

  // Epilogue: normalize; fully-masked rows (row 0, mask_k=0) output zero.
#pragma unroll
  for (int dn = 0; dn < 4; ++dn)
#pragma unroll
    for (int v = 0; v < 8; ++v) {
      int m_g = mrow0 + v;
      float val = 0.0f;
      if (rm[v] > -1e29f && rs[v] > 0.0f) val = of[dn][v] / rs[v];
      O[base + (size_t)m_g * D_ + dn * 16 + ln16] = val;
    }
}

// ---------------------------------------------------------------------------
// out = LayerNorm(x + delta) * w + b     (one 512-elem row per wave)
// ---------------------------------------------------------------------------
__global__ __launch_bounds__(256)
void add_ln_kernel(const float* __restrict__ X, const float* __restrict__ Dl,
                   const float* __restrict__ w, const float* __restrict__ b,
                   float* __restrict__ Out) {
  const int row  = blockIdx.x * 8 + (threadIdx.x >> 5);
  const int lane = threadIdx.x & 31;
  const float* xr = X + (size_t)row * D_;
  const float* dr = Dl + (size_t)row * D_;

  float hv[16];
  float sum = 0.0f;
#pragma unroll
  for (int i = 0; i < 4; ++i) {
    int c = lane * 4 + i * 128;
    f32x4 xv = *(const f32x4*)(xr + c);
    f32x4 dv = *(const f32x4*)(dr + c);
    hv[4 * i + 0] = xv.x + dv.x; hv[4 * i + 1] = xv.y + dv.y;
    hv[4 * i + 2] = xv.z + dv.z; hv[4 * i + 3] = xv.w + dv.w;
    sum += hv[4 * i + 0] + hv[4 * i + 1] + hv[4 * i + 2] + hv[4 * i + 3];
  }
#pragma unroll
  for (int s = 1; s < 32; s <<= 1) sum += __shfl_xor(sum, s, 32);
  float mu = sum * (1.0f / (float)D_);

  float vs = 0.0f;
#pragma unroll
  for (int j = 0; j < 16; ++j) { float d = hv[j] - mu; vs += d * d; }
#pragma unroll
  for (int s = 1; s < 32; s <<= 1) vs += __shfl_xor(vs, s, 32);
  float rstd = rsqrtf(vs * (1.0f / (float)D_) + 1e-5f);

#pragma unroll
  for (int i = 0; i < 4; ++i) {
    int c = lane * 4 + i * 128;
    f32x4 wv = *(const f32x4*)(w + c);
    f32x4 bv = *(const f32x4*)(b + c);
    f32x4 o;
    o.x = (hv[4 * i + 0] - mu) * rstd * wv.x + bv.x;
    o.y = (hv[4 * i + 1] - mu) * rstd * wv.y + bv.y;
    o.z = (hv[4 * i + 2] - mu) * rstd * wv.z + bv.z;
    o.w = (hv[4 * i + 3] - mu) * rstd * wv.w + bv.w;
    *(f32x4*)(Out + (size_t)row * D_ + c) = o;
  }
}

// ---------------------------------------------------------------------------
// Host-side orchestration
// ---------------------------------------------------------------------------
struct Blk {
  const float *W1, *W2, *Wk, *Wo, *Wv, *b1, *b2, *bk, *bo, *bv;
  const float *ln1_b, *ln1_w, *ln2_b, *ln2_w;
};

static void run_layer(const float* x_in, const float* v_in, int mask_k,
                      bool apply_pos, const Blk& P, int li,
                      float* ln1_out, float* final_out,
                      float* kq, float* vb, float* ao, float* po,
                      float* hb, float* fo, hipStream_t s) {
  const float* Wk = P.Wk + (size_t)li * D_ * D_;
  const float* Wv = P.Wv + (size_t)li * D_ * D_;
  const float* Wo = P.Wo + (size_t)li * D_ * D_;
  const float* W1 = P.W1 + (size_t)li * D_ * DFF_;
  const float* W2 = P.W2 + (size_t)li * DFF_ * D_;
  const float* bk = P.bk + (size_t)li * D_;
  const float* bv = P.bv + (size_t)li * D_;
  const float* bo = P.bo + (size_t)li * D_;
  const float* b1 = P.b1 + (size_t)li * DFF_;
  const float* b2 = P.b2 + (size_t)li * D_;
  const float* l1w = P.ln1_w + (size_t)li * D_;
  const float* l1b = P.ln1_b + (size_t)li * D_;
  const float* l2w = P.ln2_w + (size_t)li * D_;
  const float* l2b = P.ln2_b + (size_t)li * D_;

  dim3 blk(256);
  dim3 gD(D_ / 128, M_ / 128);      // (4, 64)
  dim3 gF(DFF_ / 128, M_ / 128);    // (16, 64)

  // Shared Q/K projection (kq_same=True, query==key_ in every call).
  gemm_bias_kernel<false><<<gD, blk, 0, s>>>(x_in, Wk, bk, kq, M_, D_, D_);
  gemm_bias_kernel<false><<<gD, blk, 0, s>>>(v_in, Wv, bv, vb, M_, D_, D_);
  attn_kernel<<<dim3(L_ / 64, B_ * H_), 128, 0, s>>>(kq, vb, ao, mask_k);
  gemm_bias_kernel<false><<<gD, blk, 0, s>>>(ao, Wo, bo, po, M_, D_, D_);
  add_ln_kernel<<<M_ / 8, 256, 0, s>>>(x_in, po, l1w, l1b, ln1_out);

  if (apply_pos) {
    gemm_bias_kernel<true><<<gF, blk, 0, s>>>(ln1_out, W1, b1, hb, M_, DFF_, D_);
    gemm_bias_kernel<false><<<gD, blk, 0, s>>>(hb, W2, b2, fo, M_, D_, DFF_);
    add_ln_kernel<<<M_ / 8, 256, 0, s>>>(ln1_out, fo, l2w, l2b, final_out);
  }
}

static Blk make_blk(void* const* d_in, int base) {
  // JAX tree flatten order (alphabetical): W1 W2 Wk Wo Wv b1 b2 bk bo bv
  //                                        ln1_b ln1_w ln2_b ln2_w
  Blk P;
  P.W1    = (const float*)d_in[base + 0];
  P.W2    = (const float*)d_in[base + 1];
  P.Wk    = (const float*)d_in[base + 2];
  P.Wo    = (const float*)d_in[base + 3];
  P.Wv    = (const float*)d_in[base + 4];
  P.b1    = (const float*)d_in[base + 5];
  P.b2    = (const float*)d_in[base + 6];
  P.bk    = (const float*)d_in[base + 7];
  P.bo    = (const float*)d_in[base + 8];
  P.bv    = (const float*)d_in[base + 9];
  P.ln1_b = (const float*)d_in[base + 10];
  P.ln1_w = (const float*)d_in[base + 11];
  P.ln2_b = (const float*)d_in[base + 12];
  P.ln2_w = (const float*)d_in[base + 13];
  return P;
}

extern "C" void kernel_launch(void* const* d_in, const int* in_sizes, int n_in,
                              void* d_out, int out_size, void* d_ws,
                              size_t ws_size, hipStream_t stream) {
  (void)in_sizes; (void)n_in; (void)out_size; (void)ws_size;

  const float* q_embed  = (const float*)d_in[0];
  const float* qa_embed = (const float*)d_in[1];
  Blk B1 = make_blk(d_in, 2);
  Blk B2 = make_blk(d_in, 16);

  // Workspace layout (floats). hb (FFN hidden, 64MB) aliases kq|vb|ao|po,
  // which are dead once LN1 has consumed po.
  const size_t SZ = (size_t)M_ * D_;        // 4,194,304 floats
  float* ws = (float*)d_ws;
  float* kq = ws + 0 * SZ;
  float* vb = ws + 1 * SZ;
  float* ao = ws + 2 * SZ;
  float* po = ws + 3 * SZ;
  float* hb = ws;                           // [M, DFF] = 4*SZ, alias of above
  float* fo = ws + 4 * SZ;                  // FFN output
  float* yb = ws + 5 * SZ;                  // y activations
  float* xb = ws + 6 * SZ;                  // x activations

  // blocks_1: y = tlayer(1, y, y, y, pos=True) twice
  run_layer(qa_embed, qa_embed, 1, true,  B1, 0, yb, yb, kq, vb, ao, po, hb, fo, stream);
  run_layer(yb,       yb,       1, true,  B1, 1, yb, yb, kq, vb, ao, po, hb, fo, stream);

  // blocks_2: alternate (1,x,x,x,no-pos) / (0,x,x,y,pos)
  run_layer(q_embed,  q_embed,  1, false, B2, 0, xb, xb, kq, vb, ao, po, hb, fo, stream);
  run_layer(xb,       yb,       0, true,  B2, 1, xb, xb, kq, vb, ao, po, hb, fo, stream);
  run_layer(xb,       xb,       1, false, B2, 2, xb, xb, kq, vb, ao, po, hb, fo, stream);
  run_layer(xb,       yb,       0, true,  B2, 3, xb, (float*)d_out,
            kq, vb, ao, po, hb, fo, stream);
}
